// PCGRModel_72292889526486
// MI455X (gfx1250) — matrix operand order repaired and checked by reading, not compiled, not run on hardware
//
#include <hip/hip_runtime.h>
#include <math.h>

#define Bb    128
#define Kc    1024
#define DM    256
#define DH    512
#define Rr    64
#define R2    128
#define CLIP  768
#define SBd   384
#define RHOc  0.35f

typedef __attribute__((ext_vector_type(16))) __bf16 v16bf;
typedef __attribute__((ext_vector_type(8)))  __bf16 v8bf;
typedef __attribute__((ext_vector_type(8)))  float  v8f;

__device__ __forceinline__ float gelu_f(float x){
  return 0.5f * x * (1.0f + erff(x * 0.70710678118654752f));
}
__device__ __forceinline__ float sigmoid_f(float x){
  return 1.0f / (1.0f + expf(-x));
}

// ---------------- t = normalize(t_clip)@Wt + bt ; v likewise --------------
__global__ void k_embed_tv(const float* __restrict__ t_clip, const float* __restrict__ Wt, const float* __restrict__ bt,
                           const float* __restrict__ v_clip, const float* __restrict__ Wv, const float* __restrict__ bv,
                           float* __restrict__ tO, float* __restrict__ vO){
  __shared__ float row[CLIP];
  __shared__ float red[256];
  int b = blockIdx.x, tid = threadIdx.x;   // 256 threads == DM
  // ---- t ----
  for (int i = tid; i < CLIP; i += 256) row[i] = t_clip[(size_t)b*CLIP + i];
  __syncthreads();
  float ss = 0.f;
  for (int i = tid; i < CLIP; i += 256) ss += row[i]*row[i];
  red[tid] = ss; __syncthreads();
  for (int s = 128; s > 0; s >>= 1){ if (tid < s) red[tid] += red[tid+s]; __syncthreads(); }
  float inv = 1.0f / fmaxf(sqrtf(red[0]), 1e-12f);
  float acc = 0.f;
  for (int i = 0; i < CLIP; ++i) acc += row[i] * Wt[(size_t)i*DM + tid];
  tO[(size_t)b*DM + tid] = acc*inv + bt[tid];
  __syncthreads();
  // ---- v ----
  for (int i = tid; i < CLIP; i += 256) row[i] = v_clip[(size_t)b*CLIP + i];
  __syncthreads();
  ss = 0.f;
  for (int i = tid; i < CLIP; i += 256) ss += row[i]*row[i];
  red[tid] = ss; __syncthreads();
  for (int s = 128; s > 0; s >>= 1){ if (tid < s) red[tid] += red[tid+s]; __syncthreads(); }
  inv = 1.0f / fmaxf(sqrtf(red[0]), 1e-12f);
  acc = 0.f;
  for (int i = 0; i < CLIP; ++i) acc += row[i] * Wv[(size_t)i*DM + tid];
  vO[(size_t)b*DM + tid] = acc*inv + bv[tid];
}

// ---------------- d = sbert@Wc + bc ; tau = sigmoid(tau_logits) ----------
__global__ void k_embed_d(const float* __restrict__ sbert, const float* __restrict__ Wc, const float* __restrict__ bc,
                          const float* __restrict__ tau_logits, float* __restrict__ dO, float* __restrict__ tauO){
  __shared__ float row[SBd];
  int k = blockIdx.x, tid = threadIdx.x;  // 256 threads
  for (int i = tid; i < SBd; i += 256) row[i] = sbert[(size_t)k*SBd + i];
  __syncthreads();
  float acc = 0.f;
  for (int i = 0; i < SBd; ++i) acc += row[i] * Wc[(size_t)i*DM + tid];
  dO[(size_t)k*DM + tid] = acc + bc[tid];
  if (tid == 0) tauO[k] = sigmoid_f(tau_logits[k]);
}

// ----- A1[br][b][j] = v@W1[0:DM] + t@W1[DM:2DM] + b1   (per-batch half) --
__global__ void k_A1(const float* __restrict__ tE, const float* __restrict__ vE,
                     const float* __restrict__ Wp1, const float* __restrict__ bp1,
                     const float* __restrict__ Wm1, const float* __restrict__ bm1,
                     float* __restrict__ A1){
  __shared__ float vr[DM], tr[DM];
  int b = blockIdx.x, br = blockIdx.y, j = threadIdx.x; // 512 threads
  if (j < DM) vr[j] = vE[(size_t)b*DM + j];
  else        tr[j-DM] = tE[(size_t)b*DM + (j-DM)];
  __syncthreads();
  const float* W1 = br ? Wm1 : Wp1;
  const float* b1 = br ? bm1 : bp1;
  float acc = b1[j];
  for (int i = 0; i < DM; ++i) acc += vr[i] * W1[(size_t)i*DH + j];
  for (int i = 0; i < DM; ++i) acc += tr[i] * W1[(size_t)(DM+i)*DH + j];
  A1[((size_t)br*Bb + b)*DH + j] = acc;
}

// ----- D1[br][k][j] = d@W1[2DM:3DM]   (per-concept half) -----------------
__global__ void k_D1(const float* __restrict__ dE,
                     const float* __restrict__ Wp1, const float* __restrict__ Wm1,
                     float* __restrict__ D1){
  __shared__ float dr[DM];
  int k = blockIdx.x, br = blockIdx.y, j = threadIdx.x; // 512 threads
  if (j < DM) dr[j] = dE[(size_t)k*DM + j];
  __syncthreads();
  const float* W1 = br ? Wm1 : Wp1;
  float acc = 0.f;
  for (int i = 0; i < DM; ++i) acc += dr[i] * W1[(size_t)(2*DM + i)*DH + j];
  D1[((size_t)br*Kc + k)*DH + j] = acc;
}

// ----- fold W2@[U|V] -> bf16, packed in WMMA B-fragment layout -----------
// frag id = (br*16 + kt)*8 + nt ; lane holds 16 contiguous bf16:
//   elem e <-> (K = kt*32 + (lane/16)*16 + e , N = nt*16 + lane%16)
__global__ void k_packW(const float* __restrict__ Wp2, const float* __restrict__ Wm2,
                        const float* __restrict__ U, const float* __restrict__ V,
                        __bf16* __restrict__ packW){
  int id = blockIdx.x, lane = threadIdx.x;          // 256 blocks x 32 lanes
  int br = id >> 7, rem = id & 127, kt = rem >> 3, nt = rem & 7;
  const float* W2 = br ? Wm2 : Wp2;
  int n  = nt*16 + (lane & 15);
  const float* UV = (n < Rr) ? U : V;
  int nc = (n < Rr) ? n : (n - Rr);
  int kb = kt*32 + (lane >> 4)*16;
  __bf16* outp = packW + ((size_t)id*32 + lane)*16;
  for (int e = 0; e < 16; ++e){
    int kk = kb + e;
    float acc = 0.f;
    for (int i = 0; i < DM; ++i) acc += W2[(size_t)kk*DM + i] * UV[(size_t)i*Rr + nc];
    outp[e] = (__bf16)acc;
  }
}

// ----- biasUV[br][n] = b2 @ [U|V] ----------------------------------------
__global__ void k_biasUV(const float* __restrict__ bp2, const float* __restrict__ bm2,
                         const float* __restrict__ U, const float* __restrict__ V,
                         float* __restrict__ biasUV){
  int tid = threadIdx.x;                            // 256 threads
  int br = tid >> 7, n = tid & 127;
  const float* b2 = br ? bm2 : bp2;
  const float* UV = (n < Rr) ? U : V;
  int nc = (n < Rr) ? n : (n - Rr);
  float acc = 0.f;
  for (int i = 0; i < DM; ++i) acc += b2[i] * UV[(size_t)i*Rr + nc];
  biasUV[tid] = acc;
}

// ----- gw[k][r] = (gelu(sbert@Wg1+bg1)@Wg2 + bg2) * w_out ---------------
__global__ void k_g(const float* __restrict__ sbert, const float* __restrict__ Wg1, const float* __restrict__ bg1,
                    const float* __restrict__ Wg2, const float* __restrict__ bg2,
                    const float* __restrict__ w_out, float* __restrict__ gw){
  __shared__ float srow[SBd];
  __shared__ float g1s[Rr];
  int k = blockIdx.x, r = threadIdx.x;              // 64 threads
  for (int i = r; i < SBd; i += 64) srow[i] = sbert[(size_t)k*SBd + i];
  __syncthreads();
  float a = bg1[r];
  for (int i = 0; i < SBd; ++i) a += srow[i] * Wg1[(size_t)i*Rr + r];
  g1s[r] = gelu_f(a);
  __syncthreads();
  float bb = bg2[r];
  for (int rr = 0; rr < Rr; ++rr) bb += g1s[rr] * Wg2[(size_t)rr*Rr + r];
  gw[(size_t)k*Rr + r] = bb * w_out[(size_t)k*Rr + r];
}

// ----- qn = normalize(concept_q) -----------------------------------------
__global__ void k_qn(const float* __restrict__ cq, float* __restrict__ qn){
  __shared__ float red[256];
  int k = blockIdx.x, tid = threadIdx.x;            // 256 threads == DM
  float x = cq[(size_t)k*DM + tid];
  red[tid] = x*x; __syncthreads();
  for (int s = 128; s > 0; s >>= 1){ if (tid < s) red[tid] += red[tid+s]; __syncthreads(); }
  float inv = 1.0f / fmaxf(sqrtf(red[0]), 1e-12f);
  qn[(size_t)k*DM + tid] = x * inv;
}

// ----- alpha[node][j] = softmax over valid parents of qn[par].qn[node] ---
// mask is provably (parents_idx != 0): all parents are > node >= 0.
__global__ void k_alpha(const float* __restrict__ qn, const int* __restrict__ pidx,
                        float* __restrict__ alpha){
  int node = blockIdx.x*128 + threadIdx.x;          // 8 x 128 = 1024
  int pj[8]; float sc[8];
  for (int j = 0; j < 8; ++j) pj[j] = pidx[node*8 + j];
  float mx = -1e30f;
  for (int j = 0; j < 8; ++j){
    float s = -1e30f;
    if (pj[j] != 0){
      s = 0.f;
      for (int i = 0; i < DM; ++i) s += qn[(size_t)pj[j]*DM + i] * qn[(size_t)node*DM + i];
    }
    sc[j] = s;
    mx = fmaxf(mx, s);
  }
  float e[8]; float sum = 0.f;
  for (int j = 0; j < 8; ++j){
    e[j] = (pj[j] != 0) ? expf(sc[j] - mx) : 0.f;
    sum += e[j];
  }
  float isum = (sum > 0.f) ? 1.f/sum : 0.f;
  for (int j = 0; j < 8; ++j) alpha[node*8 + j] = e[j]*isum;
}

// ----- main WMMA kernel: one wave per block; 16 k-rows x fixed b ---------
// Xbr = gelu(A1[br][b] + D1[br][k]) (bf16, in LDS) ; acc = X @ packW(br)
// u = tau*(accP+bP) + (1-tau)*(accM+bM) over N<64 ; v over N>=64
// logit = sum_r u*v*gw ; p = sigmoid(logit + b_out)
__global__ void __launch_bounds__(32) k_main(const float* __restrict__ A1, const float* __restrict__ D1,
                                             const __bf16* __restrict__ packW, const float* __restrict__ biasUV,
                                             const float* __restrict__ tau, const float* __restrict__ gw,
                                             const float* __restrict__ b_out, float* __restrict__ pO){
  __shared__ __bf16 sX[2][16][DH];   // 32 KB
  int lane = threadIdx.x;
  int kbase = blockIdx.x * 16;
  int b = blockIdx.y;

  // ---------------- activation fill, 8-wide per lane ----------------
  // lane owns columns j0..j0+7 for jc in {0,1}; A1 slice is loop-invariant
  // across the 16 rows -> held in registers (no LDS staging needed).
  const float* D1p = D1;
  const float* D1m = D1 + (size_t)Kc*DH;
  const float* A1p = A1 + ((size_t)0*Bb + b)*DH;
  const float* A1m = A1 + ((size_t)1*Bb + b)*DH;
  #pragma unroll
  for (int jc = 0; jc < 2; ++jc){
    int j0 = jc*256 + lane*8;
    float4 ap0 = *(const float4*)(A1p + j0);
    float4 ap1 = *(const float4*)(A1p + j0 + 4);
    float4 am0 = *(const float4*)(A1m + j0);
    float4 am1 = *(const float4*)(A1m + j0 + 4);
    for (int r = 0; r < 16; ++r){
      int k = kbase + r;
      float4 dp0 = *(const float4*)(D1p + (size_t)k*DH + j0);
      float4 dp1 = *(const float4*)(D1p + (size_t)k*DH + j0 + 4);
      float4 dm0 = *(const float4*)(D1m + (size_t)k*DH + j0);
      float4 dm1 = *(const float4*)(D1m + (size_t)k*DH + j0 + 4);
      v8bf op, om;
      op[0] = (__bf16)gelu_f(ap0.x + dp0.x);
      op[1] = (__bf16)gelu_f(ap0.y + dp0.y);
      op[2] = (__bf16)gelu_f(ap0.z + dp0.z);
      op[3] = (__bf16)gelu_f(ap0.w + dp0.w);
      op[4] = (__bf16)gelu_f(ap1.x + dp1.x);
      op[5] = (__bf16)gelu_f(ap1.y + dp1.y);
      op[6] = (__bf16)gelu_f(ap1.z + dp1.z);
      op[7] = (__bf16)gelu_f(ap1.w + dp1.w);
      om[0] = (__bf16)gelu_f(am0.x + dm0.x);
      om[1] = (__bf16)gelu_f(am0.y + dm0.y);
      om[2] = (__bf16)gelu_f(am0.z + dm0.z);
      om[3] = (__bf16)gelu_f(am0.w + dm0.w);
      om[4] = (__bf16)gelu_f(am1.x + dm1.x);
      om[5] = (__bf16)gelu_f(am1.y + dm1.y);
      om[6] = (__bf16)gelu_f(am1.z + dm1.z);
      om[7] = (__bf16)gelu_f(am1.w + dm1.w);
      *(v8bf*)&sX[0][r][j0] = op;    // one ds_store_b128
      *(v8bf*)&sX[1][r][j0] = om;    // one ds_store_b128
    }
  }
  // single wave: LDS ops from this wave complete in order before the reads below
  int m = lane & 15, hi = lane >> 4;
  v8f zero = {0.f,0.f,0.f,0.f,0.f,0.f,0.f,0.f};
  v8f accP[8], accM[8];
  #pragma unroll
  for (int nt = 0; nt < 8; ++nt){ accP[nt] = zero; accM[nt] = zero; }

  const v16bf* bfrag = (const v16bf*)packW; // [((br*16+kt)*8+nt)*32 + lane]
  for (int kt = 0; kt < 16; ++kt){
    int K0 = kt*32;
    // A fragment layout (contiguous, matches packW): lane (m,hi) holds
    // K = K0 + hi*16 + {0..15}
    v8bf pl = *(const v8bf*)&sX[0][m][K0 + hi*16];
    v8bf ph = *(const v8bf*)&sX[0][m][K0 + hi*16 + 8];
    v16bf aP = __builtin_shufflevector(pl, ph, 0,1,2,3,4,5,6,7,8,9,10,11,12,13,14,15);
    v8bf ml = *(const v8bf*)&sX[1][m][K0 + hi*16];
    v8bf mh = *(const v8bf*)&sX[1][m][K0 + hi*16 + 8];
    v16bf aM = __builtin_shufflevector(ml, mh, 0,1,2,3,4,5,6,7,8,9,10,11,12,13,14,15);
    #pragma unroll
    for (int nt = 0; nt < 8; ++nt){
      v16bf bP = bfrag[(size_t)(((0*16 + kt)*8 + nt)*32 + lane)];
      accP[nt] = __builtin_amdgcn_wmma_f32_16x16x32_bf16(false, aP, false, bP, (short)0, accP[nt], false, false);
      v16bf bM = bfrag[(size_t)(((1*16 + kt)*8 + nt)*32 + lane)];
      accM[nt] = __builtin_amdgcn_wmma_f32_16x16x32_bf16(false, aM, false, bM, (short)0, accM[nt], false, false);
    }
  }
  // epilogue: C/D layout: vgpr v, lane -> (M = v + 8*hi, N = lane%16)
  int nl = m;
  float s[8], tauv[8];
  #pragma unroll
  for (int v = 0; v < 8; ++v){ tauv[v] = tau[kbase + v + 8*hi]; s[v] = 0.f; }
  float bP_[8], bM_[8];
  #pragma unroll
  for (int nt = 0; nt < 8; ++nt){ bP_[nt] = biasUV[nt*16 + nl]; bM_[nt] = biasUV[R2 + nt*16 + nl]; }
  #pragma unroll
  for (int v = 0; v < 8; ++v){
    int krow = kbase + v + 8*hi;
    float tv = tauv[v], omtv = 1.f - tauv[v];
    #pragma unroll
    for (int nt = 0; nt < 4; ++nt){
      float u = tv*(accP[nt][v]   + bP_[nt])   + omtv*(accM[nt][v]   + bM_[nt]);
      float w = tv*(accP[nt+4][v] + bP_[nt+4]) + omtv*(accM[nt+4][v] + bM_[nt+4]);
      s[v] += u * w * gw[(size_t)krow*Rr + nt*16 + nl];
    }
  }
  #pragma unroll
  for (int v = 0; v < 8; ++v){
    s[v] += __shfl_xor(s[v], 1, 32);
    s[v] += __shfl_xor(s[v], 2, 32);
    s[v] += __shfl_xor(s[v], 4, 32);
    s[v] += __shfl_xor(s[v], 8, 32);
  }
  if (nl == 0){
    #pragma unroll
    for (int v = 0; v < 8; ++v){
      int krow = kbase + v + 8*hi;
      pO[(size_t)b*Kc + krow] = sigmoid_f(s[v] + b_out[krow]);
    }
  }
}

// ----- sequential DAG propagation; batch columns independent -------------
__global__ void k_prop(const float* __restrict__ pIn, const float* __restrict__ alpha,
                       const int* __restrict__ pidx, float* __restrict__ out){
  __shared__ float pc[Kc][8];  // 32 KB, conflict-free: [k][b_local]
  int tid = threadIdx.x;                           // 0..7
  int b = blockIdx.x*8 + tid;
  for (int k = 0; k < Kc; ++k) pc[k][tid] = pIn[(size_t)b*Kc + k];
  for (int node = Kc - 1; node >= 0; --node){
    int p0 = pidx[node*8 + 0];
    float prod = 1.f;
    #pragma unroll
    for (int j = 0; j < 8; ++j){
      int pj = pidx[node*8 + j];
      if (pj != 0) prod *= alpha[node*8 + j] * pc[pj][tid];
    }
    if (p0 != 0) pc[node][tid] = RHOc * pc[node][tid] + (1.f - RHOc) * prod;
  }
  for (int k = 0; k < Kc; ++k) out[(size_t)b*Kc + k] = pc[k][tid];
}

extern "C" void kernel_launch(void* const* d_in, const int* in_sizes, int n_in,
                              void* d_out, int out_size, void* d_ws, size_t ws_size,
                              hipStream_t stream){
  (void)in_sizes; (void)n_in; (void)out_size; (void)ws_size;
  const float* t_clip = (const float*)d_in[0];
  const float* v_clip = (const float*)d_in[1];
  const float* sbert  = (const float*)d_in[2];
  const float* Wt  = (const float*)d_in[3];
  const float* bt  = (const float*)d_in[4];
  const float* Wv  = (const float*)d_in[5];
  const float* bv  = (const float*)d_in[6];
  const float* Wc  = (const float*)d_in[7];
  const float* bc  = (const float*)d_in[8];
  const float* Wp1 = (const float*)d_in[9];
  const float* bp1 = (const float*)d_in[10];
  const float* Wp2 = (const float*)d_in[11];
  const float* bp2 = (const float*)d_in[12];
  const float* Wm1 = (const float*)d_in[13];
  const float* bm1 = (const float*)d_in[14];
  const float* Wm2 = (const float*)d_in[15];
  const float* bm2 = (const float*)d_in[16];
  const float* tau_logits = (const float*)d_in[17];
  const float* U   = (const float*)d_in[18];
  const float* V   = (const float*)d_in[19];
  const float* Wg1 = (const float*)d_in[20];
  const float* bg1 = (const float*)d_in[21];
  const float* Wg2 = (const float*)d_in[22];
  const float* bg2 = (const float*)d_in[23];
  const float* w_out = (const float*)d_in[24];
  const float* b_out = (const float*)d_in[25];
  const float* cq    = (const float*)d_in[26];
  const int*   pidx  = (const int*)d_in[27];
  float* out = (float*)d_out;

  char* ws = (char*)d_ws;
  size_t off = 0;
  auto alloc = [&](size_t bytes)->char*{
    char* p = ws + off; off = (off + bytes + 255) & ~(size_t)255; return p;
  };
  float*  tE     = (float*) alloc((size_t)Bb*DM*4);
  float*  vE     = (float*) alloc((size_t)Bb*DM*4);
  float*  dE     = (float*) alloc((size_t)Kc*DM*4);
  float*  tauW   = (float*) alloc((size_t)Kc*4);
  float*  A1     = (float*) alloc((size_t)2*Bb*DH*4);
  float*  D1     = (float*) alloc((size_t)2*Kc*DH*4);
  float*  biasUV = (float*) alloc((size_t)2*R2*4);
  __bf16* packW  = (__bf16*)alloc((size_t)2*16*8*32*16*2);
  float*  gw     = (float*) alloc((size_t)Kc*Rr*4);
  float*  qn     = (float*) alloc((size_t)Kc*DM*4);
  float*  alphaW = (float*) alloc((size_t)Kc*8*4);
  float*  pW     = (float*) alloc((size_t)Bb*Kc*4);

  k_embed_tv<<<Bb, 256, 0, stream>>>(t_clip, Wt, bt, v_clip, Wv, bv, tE, vE);
  k_embed_d <<<Kc, 256, 0, stream>>>(sbert, Wc, bc, tau_logits, dE, tauW);
  k_A1<<<dim3(Bb,2), DH, 0, stream>>>(tE, vE, Wp1, bp1, Wm1, bm1, A1);
  k_D1<<<dim3(Kc,2), DH, 0, stream>>>(dE, Wp1, Wm1, D1);
  k_packW<<<256, 32, 0, stream>>>(Wp2, Wm2, U, V, packW);
  k_biasUV<<<1, 256, 0, stream>>>(bp2, bm2, U, V, biasUV);
  k_g<<<Kc, 64, 0, stream>>>(sbert, Wg1, bg1, Wg2, bg2, w_out, gw);
  k_qn<<<Kc, 256, 0, stream>>>(cq, qn);
  k_alpha<<<8, 128, 0, stream>>>(qn, pidx, alphaW);
  k_main<<<dim3(Kc/16, Bb), 32, 0, stream>>>(A1, D1, packW, biasUV, tauW, gw, b_out, pW);
  k_prop<<<16, 8, 0, stream>>>(pW, alphaW, pidx, out);
}
